// Encoder_60498909332042
// MI455X (gfx1250) — matrix operand order repaired and checked by reading, not compiled, not run on hardware
//
#include <hip/hip_runtime.h>
#include <hip/hip_bf16.h>

// Problem constants (from reference): B=512, T-1=256, N=128, H=128
#define B_  512
#define T_  256
#define N_  128
#define H_  128
#define G4  512   // 4*H gate width

typedef __attribute__((ext_vector_type(16))) __bf16 v16bf;
typedef __attribute__((ext_vector_type(8)))  float  v8f;

union BF16x16 { v16bf v; uint4 q[2]; };

__device__ __forceinline__ unsigned short f2bf(float f) {
    unsigned u = __builtin_bit_cast(unsigned, f);
    u += 0x7FFFu + ((u >> 16) & 1u);   // round to nearest even
    return (unsigned short)(u >> 16);
}
__device__ __forceinline__ float sigf(float x) { return 1.0f / (1.0f + __expf(-x)); }

// ---------------------------------------------------------------------------
// Kernel 1: per-batch attention weights (softmax over N) + input_weighted.
// alpha[b,n] = softmax_n( x_series[b,n,:] . attn_W[2H:] )   (h/c/bias cancel,
// so alpha is independent of t and of the LSTM state -> compute once).
// Writes input_weighted (fp32, first half of d_out) and x_tilde in bf16 (ws).
// ---------------------------------------------------------------------------
__global__ void attn_softmax_kernel(const float* __restrict__ x,
                                    const float* __restrict__ attn_W,
                                    float* __restrict__ iw_out,
                                    unsigned short* __restrict__ xt_bf)
{
    __shared__ float red[N_];
    const int b = blockIdx.x, n = threadIdx.x;
    const float* xb = x + (size_t)b * T_ * N_;

    float acc = 0.f;
    #pragma unroll 4
    for (int t = 0; t < T_; ++t)
        acc += xb[t * N_ + n] * attn_W[2 * H_ + t];

    red[n] = acc; __syncthreads();
    for (int s = 64; s > 0; s >>= 1) {
        if (n < s) red[n] = fmaxf(red[n], red[n + s]);
        __syncthreads();
    }
    const float m = red[0]; __syncthreads();
    const float e = __expf(acc - m);
    red[n] = e; __syncthreads();
    for (int s = 64; s > 0; s >>= 1) {
        if (n < s) red[n] += red[n + s];
        __syncthreads();
    }
    const float alpha = e / red[0];

    for (int t = 0; t < T_; ++t) {
        const float v = alpha * xb[t * N_ + n];
        const size_t o = (size_t)b * T_ * N_ + (size_t)t * N_ + n;
        __builtin_nontemporal_store(v, iw_out + o);   // output, never re-read
        xt_bf[o] = f2bf(v);                           // re-read soon: keep in L2
    }
}

// ---------------------------------------------------------------------------
// Kernel 2: convert W_ih and W_hh to bf16 (once).
// ---------------------------------------------------------------------------
__global__ void w2bf_kernel(const float* __restrict__ wih, const float* __restrict__ whh,
                            unsigned short* __restrict__ wih_bf, unsigned short* __restrict__ whh_bf)
{
    int i = blockIdx.x * blockDim.x + threadIdx.x;
    if (i < G4 * N_) wih_bf[i] = f2bf(wih[i]);
    i -= G4 * N_;
    if (i >= 0 && i < G4 * H_) whh_bf[i] = f2bf(whh[i]);
}

// ---------------------------------------------------------------------------
// Kernel 3: time-parallel input GEMM.
// gates_x[(b*T+t), 0:512] = x_tilde[b,t,:] @ W_ih^T + (b_ih + b_hh)
// M = B*T = 131072, N = 512, K = 128.  Block = 8 waves, 16 rows each.
// ---------------------------------------------------------------------------
__global__ void __launch_bounds__(256)
xgemm_kernel(const unsigned short* __restrict__ xt_bf,
             const unsigned short* __restrict__ wih_bf,
             const float* __restrict__ b_ih,
             const float* __restrict__ b_hh,
             float* __restrict__ gates_x)
{
    const int wave = threadIdx.x >> 5, lane = threadIdx.x & 31;
    const int lr = lane & 15, hi = lane >> 4;
    const int rowBase = blockIdx.x * 128 + wave * 16;

    // A fragments: 16x32 bf16 tiles of x_tilde rows, K = 0..127 in 4 tiles
    v16bf A[4];
    #pragma unroll
    for (int kt = 0; kt < 4; ++kt) {
        const int kb = kt * 32 + hi * 8;
        BF16x16 u;
        u.q[0] = *(const uint4*)(xt_bf + (size_t)(rowBase + lr) * N_ + kb);
        u.q[1] = *(const uint4*)(xt_bf + (size_t)(rowBase + lr) * N_ + kb + 16);
        A[kt] = u.v;
    }

    for (int nt = 0; nt < 32; ++nt) {
        const int col  = nt * 16 + lr;
        const float bias = b_ih[col] + b_hh[col];   // same for every M in the tile
        v8f c = {bias, bias, bias, bias, bias, bias, bias, bias};
        #pragma unroll
        for (int kt = 0; kt < 4; ++kt) {
            const int k = kt * 32 + hi * 8;
            BF16x16 bu;
            bu.q[0] = *(const uint4*)(wih_bf + col * N_ + k);
            bu.q[1] = *(const uint4*)(wih_bf + col * N_ + k + 16);
            c = __builtin_amdgcn_wmma_f32_16x16x32_bf16(false, A[kt], false, bu.v,
                                                        (short)0, c, false, false);
        }
        #pragma unroll
        for (int v = 0; v < 8; ++v)   // 268 MB stream: don't thrash L2
            __builtin_nontemporal_store(
                c[v], gates_x + (size_t)(rowBase + v + hi * 8) * G4 + col);
    }
}

// ---------------------------------------------------------------------------
// Kernel 4: sequential LSTM.  Recurrence is independent per batch row:
// 32 blocks each own 16 batch rows and loop t = 0..255 with only LDS barriers.
// Wave w owns gate columns {16w..16w+15} of ALL FOUR gates, so i/f/g/o of a
// cell land in the same lane -> activations + c-state are pure register math.
// W_hh bf16 B-fragments persist in registers across all 256 steps; only the
// bf16 h vector round-trips LDS (A-fragment transpose for the next step).
// ---------------------------------------------------------------------------
__global__ void __launch_bounds__(256)
lstm_seq_kernel(const float* __restrict__ gates_x,
                const unsigned short* __restrict__ whh_bf,
                float* __restrict__ ie_out)
{
    __shared__ unsigned short h_lds[16 * H_];   // 4 KB: h state (bf16)

    const int tid  = threadIdx.x;
    const int wave = tid >> 5, lane = tid & 31;
    const int lr = lane & 15, hi = lane >> 4;
    const int bBase = blockIdx.x * 16;   // batch rows owned by this block
    const int colw  = wave * 16;         // hidden-column group owned by wave

    #pragma unroll
    for (int q = 0; q < 8; ++q) h_lds[q * 256 + tid] = 0;

    // Persistent W_hh B-fragments: gate g, K-tile kt; column n = g*128+colw+lr
    v16bf Bfr[16];
    #pragma unroll
    for (int g = 0; g < 4; ++g)
        #pragma unroll
        for (int kt = 0; kt < 4; ++kt) {
            const int n = g * 128 + colw + lr;
            const int k = kt * 32 + hi * 8;
            BF16x16 u;
            u.q[0] = *(const uint4*)(whh_bf + n * H_ + k);
            u.q[1] = *(const uint4*)(whh_bf + n * H_ + k + 16);
            Bfr[g * 4 + kt] = u.v;
        }

    float creg[8];                      // lane-private c state: (M=v+8*hi, colw+lr)
    #pragma unroll
    for (int v = 0; v < 8; ++v) creg[v] = 0.f;

    __syncthreads();

    for (int t = 0; t < T_; ++t) {
        // x-side gate pre-activations: overlap these loads with the WMMA chain
        float Gx[4][8];
        #pragma unroll
        for (int g = 0; g < 4; ++g)
            #pragma unroll
            for (int v = 0; v < 8; ++v)
                Gx[g][v] = __builtin_nontemporal_load(
                    gates_x + ((size_t)(bBase + v + 8 * hi) * T_ + t) * G4
                            + g * 128 + colw + lr);

        // A fragments of h from LDS (bf16, 16x32 layout)
        v16bf A[4];
        #pragma unroll
        for (int kt = 0; kt < 4; ++kt) {
            const int kb = kt * 32 + hi * 8;
            BF16x16 u;
            u.q[0] = *(const uint4*)&h_lds[lr * H_ + kb];
            u.q[1] = *(const uint4*)&h_lds[lr * H_ + kb + 16];
            A[kt] = u.v;
        }

        v8f C[4];
        #pragma unroll
        for (int g = 0; g < 4; ++g) C[g] = (v8f){0, 0, 0, 0, 0, 0, 0, 0};
        #pragma unroll
        for (int kt = 0; kt < 4; ++kt)
            #pragma unroll
            for (int g = 0; g < 4; ++g)
                C[g] = __builtin_amdgcn_wmma_f32_16x16x32_bf16(false, A[kt], false,
                           Bfr[g * 4 + kt], (short)0, C[g], false, false);

        __syncthreads();   // everyone finished reading h_lds for this step

        // Lane-private gate activations + state update (i,f,g,o gate order)
        #pragma unroll
        for (int v = 0; v < 8; ++v) {
            const float i_ = sigf (C[0][v] + Gx[0][v]);
            const float f_ = sigf (C[1][v] + Gx[1][v]);
            const float g_ = tanhf(C[2][v] + Gx[2][v]);
            const float o_ = sigf (C[3][v] + Gx[3][v]);
            const float cn = f_ * creg[v] + i_ * g_;
            creg[v] = cn;
            const float hn = o_ * tanhf(cn);
            __builtin_nontemporal_store(hn,
                ie_out + ((size_t)(bBase + v + 8 * hi) * T_ + t) * H_ + colw + lr);
            h_lds[(v + 8 * hi) * H_ + colw + lr] = f2bf(hn);
        }
        __syncthreads();   // h_lds ready for next step's A fragments
    }
}

// ---------------------------------------------------------------------------
extern "C" void kernel_launch(void* const* d_in, const int* in_sizes, int n_in,
                              void* d_out, int out_size, void* d_ws, size_t ws_size,
                              hipStream_t stream)
{
    (void)in_sizes; (void)n_in; (void)out_size; (void)ws_size;
    const float* x      = (const float*)d_in[0];
    const float* attn_W = (const float*)d_in[1];
    // d_in[2] = attn_b: cancels in the softmax, unused.
    const float* W_ih   = (const float*)d_in[3];
    const float* W_hh   = (const float*)d_in[4];
    const float* b_ih   = (const float*)d_in[5];
    const float* b_hh   = (const float*)d_in[6];

    float* iw_out = (float*)d_out;                               // (B,T,N)
    float* ie_out = (float*)d_out + (size_t)B_ * T_ * N_;        // (B,T,H)

    char* ws = (char*)d_ws;
    unsigned short* xt_bf  = (unsigned short*)ws;                         // 33.5 MB
    unsigned short* wih_bf = (unsigned short*)(ws + 33554432);            // 128 KB
    unsigned short* whh_bf = (unsigned short*)(ws + 33554432 + 131072);   // 128 KB
    float*          gates_x = (float*)(ws + 33554432 + 262144);          // 268 MB

    attn_softmax_kernel<<<B_, N_, 0, stream>>>(x, attn_W, iw_out, xt_bf);
    w2bf_kernel<<<(2 * G4 * N_) / 256, 256, 0, stream>>>(W_ih, W_hh, wih_bf, whh_bf);
    xgemm_kernel<<<(B_ * T_) / 128, 256, 0, stream>>>(xt_bf, wih_bf, b_ih, b_hh, gates_x);
    lstm_seq_kernel<<<B_ / 16, 256, 0, stream>>>(gates_x, whh_bf, ie_out);
}
